// MambaBlock_11433202942059
// MI455X (gfx1250) — matrix operand order repaired and checked by reading, not compiled, not run on hardware
//
#include <hip/hip_runtime.h>

// ---------------- problem constants (from reference) ----------------
#define D_MODEL 1024
#define CCH     2048          // inner channels C
#define STATE   16
#define RANK    64
#define KCONV   4
#define LSEQ    2048
#define NPROJ   (RANK + 2*STATE)   // 96

typedef __attribute__((ext_vector_type(16))) __bf16 v16bf;
typedef __attribute__((ext_vector_type(8)))  __bf16 v8bf;
typedef __attribute__((ext_vector_type(8)))  float  v8f;

__device__ __forceinline__ float softplus_f(float v) {
    return v > 20.0f ? v : log1pf(__expf(v));
}
__device__ __forceinline__ float silu_f(float v) {
    return v / (1.0f + __expf(-v));
}

// ---------------- fp32 -> bf16 convert ----------------
__global__ void cvt_bf16_kernel(const float* __restrict__ src, __bf16* __restrict__ dst, int n) {
    int i = blockIdx.x * blockDim.x + threadIdx.x;
    if (i < n) dst[i] = (__bf16)src[i];
}

// pack dlt (first 64 cols of proj, ld=96) into dense bf16 [L,64]
__global__ void pack_dlt_kernel(const float* __restrict__ proj, __bf16* __restrict__ dltb) {
    int i = blockIdx.x * blockDim.x + threadIdx.x;      // L*64
    if (i >= LSEQ * RANK) return;
    int t = i >> 6, r = i & 63;
    dltb[i] = (__bf16)proj[t * NPROJ + r];
}

// ---------------- WMMA bf16 GEMM:  C[M,N] = A[M,K] @ W[N,K]^T ----------------
// Each wave computes a (TM*16) x (TN*16) register-blocked output tile.
// TM=4,TN=2: 8 accumulators (64 VGPRs) + 6 fragments (48 VGPRs, x2 for load
// pipelining) -> ~180 VGPRs total, fits the budget WITHOUT spilling (the 4x4
// variant spilled accumulators to scratch). 8 WMMAs per 6 fragment loads
// -> ~22 FLOP/byte from L2 and 8 independent accumulation chains.
// 8 waves (256 threads) per block; tile assignment is wave-uniform so EXEC
// stays all-1s for WMMA.
// EPI: 0 = plain fp32 store, 1 = softplus(acc + bias[n])
template <int EPI, int TM, int TN>
__global__ void __launch_bounds__(256, 1)
gemm_wmma_bf16(const __bf16* __restrict__ A,
               const __bf16* __restrict__ W,
               float* __restrict__ Cout,
               const float* __restrict__ bias,
               int M, int N, int K,
               int lda, int ldb, int ldc)
{
    const int wave = threadIdx.x >> 5;
    const int lane = threadIdx.x & 31;
    const int nT2 = N / (16 * TN);
    const int mT2 = M / (16 * TM);
    const int tile = blockIdx.x * 8 + wave;
    if (tile >= mT2 * nT2) return;                  // wave-uniform exit
    const int m0 = (tile / nT2) * (16 * TM);
    const int n0 = (tile % nT2) * (16 * TN);
    const int half = lane >> 4;                     // wave32 half
    const int mn   = lane & 15;

    v8f acc[TM][TN];
#pragma unroll
    for (int mt = 0; mt < TM; ++mt)
#pragma unroll
        for (int nt = 0; nt < TN; ++nt) acc[mt][nt] = (v8f){};

    const __bf16* aBase[TM];
    const __bf16* bBase[TN];
#pragma unroll
    for (int mt = 0; mt < TM; ++mt)
        aBase[mt] = A + (size_t)(m0 + mt * 16 + mn) * lda + half * 8;
#pragma unroll
    for (int nt = 0; nt < TN; ++nt)
        bBase[nt] = W + (size_t)(n0 + nt * 16 + mn) * ldb + half * 8;

    for (int k0 = 0; k0 < K; k0 += 32) {
#pragma unroll
        for (int mt = 0; mt < TM; ++mt) __builtin_prefetch(aBase[mt] + k0 + 64, 0, 3);
#pragma unroll
        for (int nt = 0; nt < TN; ++nt) __builtin_prefetch(bBase[nt] + k0 + 64, 0, 3);

        v16bf af[TM], bq[TN];
#pragma unroll
        for (int mt = 0; mt < TM; ++mt) {
            // A frag: lane m; elements 0..7 -> k=k0+half*8+i, 8..15 -> k0+16+half*8+i
            v8bf a_lo = *(const v8bf*)(aBase[mt] + k0);
            v8bf a_hi = *(const v8bf*)(aBase[mt] + k0 + 16);
#pragma unroll
            for (int i = 0; i < 8; ++i) { af[mt][i] = a_lo[i]; af[mt][i + 8] = a_hi[i]; }
        }
#pragma unroll
        for (int nt = 0; nt < TN; ++nt) {
            // B frag: lane n; elements 0..7 -> k=k0+half*8+i, 8..15 -> k0+16+half*8+i
            v8bf b_lo = *(const v8bf*)(bBase[nt] + k0);
            v8bf b_hi = *(const v8bf*)(bBase[nt] + k0 + 16);
#pragma unroll
            for (int i = 0; i < 8; ++i) { bq[nt][i] = b_lo[i]; bq[nt][i + 8] = b_hi[i]; }
        }
#pragma unroll
        for (int mt = 0; mt < TM; ++mt)
#pragma unroll
            for (int nt = 0; nt < TN; ++nt)
                acc[mt][nt] = __builtin_amdgcn_wmma_f32_16x16x32_bf16(
                                  false, af[mt], false, bq[nt], (short)0,
                                  acc[mt][nt], false, false);
    }

#pragma unroll
    for (int nt = 0; nt < TN; ++nt) {
        const int n = n0 + nt * 16 + mn;
        float bv = 0.0f;
        if (EPI == 1) bv = bias[n];
#pragma unroll
        for (int mt = 0; mt < TM; ++mt) {
#pragma unroll
            for (int r = 0; r < 8; ++r) {           // D vgpr r -> row +half*8+r
                float v = acc[mt][nt][r];
                if (EPI == 1) v = softplus_f(v + bv);
                Cout[(size_t)(m0 + mt * 16 + half * 8 + r) * ldc + n] = v;
            }
        }
    }
}

// ---------------- causal depthwise conv (K=4) + bias + SiLU ----------------
// xp lives in xw[:, 0:2048] (row stride 2C = 4096)
__global__ void conv_silu_kernel(const float* __restrict__ xw,
                                 const float* __restrict__ conv_w,
                                 const float* __restrict__ conv_b,
                                 float* __restrict__ xc, __bf16* __restrict__ xcb)
{
    int idx = blockIdx.x * blockDim.x + threadIdx.x;   // t*C + c
    if (idx >= LSEQ * CCH) return;
    int c = idx & (CCH - 1);
    int t = idx >> 11;
    float acc = conv_b[c];
#pragma unroll
    for (int j = 0; j < KCONV; ++j) {
        int tt = t - (KCONV - 1) + j;
        if (tt >= 0) acc += conv_w[c * KCONV + j] * xw[(size_t)tt * (2 * CCH) + c];
    }
    float s = silu_f(acc);
    xc[idx]  = s;
    xcb[idx] = (__bf16)s;
}

// ---------------- selective scan (sequential in t, parallel over c,s) ----------------
// thread layout: lane = s + 16*(c_local&1); 16 channels per 256-thread block
__global__ void scan_kernel(const float* __restrict__ delta,
                            const float* __restrict__ xc,
                            const float* __restrict__ proj,
                            const float* __restrict__ xw,     // res at col 2048+
                            const float* __restrict__ A_log,
                            const float* __restrict__ Dp,
                            __bf16* __restrict__ ygb)
{
    const int s  = threadIdx.x & 15;
    const int cl = threadIdx.x >> 4;
    const int c  = blockIdx.x * 16 + cl;
    const float eA = __expf(A_log[c * STATE + s]);
    const float Dv = Dp[c];
    float h = 0.0f;
    for (int t = 0; t < LSEQ; ++t) {
        float dl = delta[(size_t)t * CCH + c];
        float xv = xc[(size_t)t * CCH + c];
        float bm = proj[t * NPROJ + RANK + s];
        float cm = proj[t * NPROJ + RANK + STATE + s];
        if (t > 0) h = __expf(-dl * eA) * h + dl * bm * xv;   // h0 = 0, B_0 unused
        float contrib = h * cm;
        contrib += __shfl_xor(contrib, 8, 16);
        contrib += __shfl_xor(contrib, 4, 16);
        contrib += __shfl_xor(contrib, 2, 16);
        contrib += __shfl_xor(contrib, 1, 16);
        if (s == 0) {
            float y  = contrib + Dv * xv;
            float r  = xw[(size_t)t * (2 * CCH) + CCH + c];
            float yg = y * silu_f(r);
            ygb[(size_t)t * CCH + c] = (__bf16)yg;
        }
    }
}

// ---------------- host-side launch ----------------
static inline size_t rnd256(size_t b) { return (b + 255) & ~(size_t)255; }

extern "C" void kernel_launch(void* const* d_in, const int* in_sizes, int n_in,
                              void* d_out, int out_size, void* d_ws, size_t ws_size,
                              hipStream_t stream) {
    const float* x       = (const float*)d_in[0];   // [L, D]
    const float* W_in    = (const float*)d_in[1];   // [2C, D]
    const float* W_out   = (const float*)d_in[2];   // [D, C]
    const float* conv_w  = (const float*)d_in[3];   // [C,1,K]
    const float* conv_b  = (const float*)d_in[4];   // [C]
    const float* A_log   = (const float*)d_in[5];   // [C, S]
    const float* D_param = (const float*)d_in[6];   // [C]
    const float* W_x     = (const float*)d_in[7];   // [96, C]
    const float* W_delta = (const float*)d_in[8];   // [C, 64]
    const float* b_delta = (const float*)d_in[9];   // [C]
    float* out = (float*)d_out;

    char* ws = (char*)d_ws;
    size_t off = 0;
    auto take = [&](size_t bytes) { char* p = ws + off; off += rnd256(bytes); return p; };

    __bf16* xb    = (__bf16*)take((size_t)LSEQ * D_MODEL * 2);
    __bf16* Winb  = (__bf16*)take((size_t)2 * CCH * D_MODEL * 2);
    __bf16* Wxb   = (__bf16*)take((size_t)NPROJ * CCH * 2);
    __bf16* Wdb   = (__bf16*)take((size_t)CCH * RANK * 2);
    __bf16* Woutb = (__bf16*)take((size_t)D_MODEL * CCH * 2);
    float*  xw    = (float*)take((size_t)LSEQ * 2 * CCH * 4);   // [xp | res]
    float*  xc    = (float*)take((size_t)LSEQ * CCH * 4);
    __bf16* xcb   = (__bf16*)take((size_t)LSEQ * CCH * 2);
    float*  proj  = (float*)take((size_t)LSEQ * NPROJ * 4);
    __bf16* dltb  = (__bf16*)take((size_t)LSEQ * RANK * 2);
    float*  delta = (float*)take((size_t)LSEQ * CCH * 4);
    __bf16* ygb   = (__bf16*)take((size_t)LSEQ * CCH * 2);

    const int TPB = 256;
    auto blocks = [](long n, int tpb) { return (int)((n + tpb - 1) / tpb); };

    // 1) converts to bf16
    cvt_bf16_kernel<<<blocks((long)LSEQ * D_MODEL, TPB), TPB, 0, stream>>>(x, xb, LSEQ * D_MODEL);
    cvt_bf16_kernel<<<blocks((long)2 * CCH * D_MODEL, TPB), TPB, 0, stream>>>(W_in, Winb, 2 * CCH * D_MODEL);
    cvt_bf16_kernel<<<blocks((long)NPROJ * CCH, TPB), TPB, 0, stream>>>(W_x, Wxb, NPROJ * CCH);
    cvt_bf16_kernel<<<blocks((long)CCH * RANK, TPB), TPB, 0, stream>>>(W_delta, Wdb, CCH * RANK);
    cvt_bf16_kernel<<<blocks((long)D_MODEL * CCH, TPB), TPB, 0, stream>>>(W_out, Woutb, D_MODEL * CCH);

    // 2) xw = x @ W_in^T   [2048 x 4096], K=1024  (64x32 per wave)
    {
        int waves = (LSEQ / 64) * ((2 * CCH) / 32);
        gemm_wmma_bf16<0, 4, 2><<<(waves + 7) / 8, TPB, 0, stream>>>(
            xb, Winb, xw, nullptr, LSEQ, 2 * CCH, D_MODEL, D_MODEL, D_MODEL, 2 * CCH);
    }

    // 3) causal depthwise conv + SiLU -> xc (fp32 + bf16)
    conv_silu_kernel<<<blocks((long)LSEQ * CCH, TPB), TPB, 0, stream>>>(xw, conv_w, conv_b, xc, xcb);

    // 4) proj = xc @ W_x^T   [2048 x 96], K=2048  (64x16 per wave)
    {
        int waves = (LSEQ / 64) * (NPROJ / 16);
        gemm_wmma_bf16<0, 4, 1><<<(waves + 7) / 8, TPB, 0, stream>>>(
            xcb, Wxb, proj, nullptr, LSEQ, NPROJ, CCH, CCH, CCH, NPROJ);
    }

    // 5) delta = softplus(dlt @ W_delta^T + b_delta)   [2048 x 2048], K=64
    pack_dlt_kernel<<<blocks((long)LSEQ * RANK, TPB), TPB, 0, stream>>>(proj, dltb);
    {
        int waves = (LSEQ / 64) * (CCH / 32);
        gemm_wmma_bf16<1, 4, 2><<<(waves + 7) / 8, TPB, 0, stream>>>(
            dltb, Wdb, delta, b_delta, LSEQ, CCH, RANK, RANK, RANK, CCH);
    }

    // 6) selective scan + gating -> ygb (bf16)
    scan_kernel<<<CCH / 16, TPB, 0, stream>>>(delta, xc, proj, xw, A_log, D_param, ygb);

    // 7) out = yg @ W_out^T   [2048 x 1024], K=2048  (64x32 per wave)
    {
        int waves = (LSEQ / 64) * (D_MODEL / 32);
        gemm_wmma_bf16<0, 4, 2><<<(waves + 7) / 8, TPB, 0, stream>>>(
            ygb, Woutb, out, nullptr, LSEQ, D_MODEL, CCH, CCH, CCH, D_MODEL);
    }
    (void)in_sizes; (void)n_in; (void)out_size; (void)ws_size;
}
